// GNN_29978871726168
// MI455X (gfx1250) — matrix operand order repaired and checked by reading, not compiled, not run on hardware
//
#include <hip/hip_runtime.h>
#include <math.h>

typedef _Float16 half_t;
typedef __attribute__((ext_vector_type(16))) _Float16 v16h;
typedef __attribute__((ext_vector_type(8)))  float    v8f;

#define N_NODES 5000
#define T_STEPS 100
#define INDIM   32
#define FEAT    64
#define GATES   256   // 4*FEAT
#define HIDD    64
#define OUTD    8
#define HEADS   4
#define NEDGE   160000
#define FIN     72    // FEAT + OUTD
#define FINP    96    // padded row stride / K padded to 3x32
#define QKV     256   // HEADS*HIDD
#define NTILES  ((N_NODES + 15) / 16)   // 313

__device__ __forceinline__ float fast_tanh(float x) {
#if __has_builtin(__builtin_amdgcn_tanhf)
  return __builtin_amdgcn_tanhf(x);
#else
  return tanhf(x);
#endif
}
__device__ __forceinline__ float sigmf(float x) { return 0.5f * fast_tanh(0.5f * x) + 0.5f; }

__device__ __forceinline__ void atomicMaxF(float* addr, float val) {
  int* ia = (int*)addr;
  int old = *ia;
  while (__int_as_float(old) < val) {
    int assumed = old;
    old = atomicCAS(ia, assumed, __float_as_int(val));
    if (old == assumed) break;
  }
}

// ---------------- WMMA fragment helpers (wave32, f16 16x16x32) -------------
// A fragment directly from row-major f32 source. ISA layout decomposes into
// two contiguous 8-element runs per lane: k = k0+8g+[0,8) and k0+16+8g+[0,8).
__device__ __forceinline__ v16h load_a_rows(const float* __restrict__ rowp, bool valid)
{
  v16h a;
  if (valid) {
#pragma unroll
    for (int i = 0; i < 8; ++i) a[i] = (half_t)rowp[i];
#pragma unroll
    for (int i = 0; i < 8; ++i) a[8 + i] = (half_t)rowp[16 + i];
  } else {
#pragma unroll
    for (int i = 0; i < 16; ++i) a[i] = (half_t)0.f;
  }
  return a;
}

// B fragment pre-packed contiguously per lane: frag[((kt*ntiles+nt)*32+lane)*16]
__device__ __forceinline__ v16h load_b_frag(const half_t* __restrict__ frag,
                                            int ntiles, int kt, int nt, int lane)
{
  const half_t* p = frag + ((((size_t)kt * ntiles + nt) * 32 + lane) << 4);
  return *(const v16h*)p;
}

// ---------------- setup kernels --------------------------------------------
__global__ void init_state_kernel(float* __restrict__ h, float* __restrict__ c,
                                  float* __restrict__ theta, float* __restrict__ hin)
{
  size_t idx = (size_t)blockIdx.x * blockDim.x + threadIdx.x;
  if (idx < (size_t)N_NODES * FEAT) { h[idx] = 0.f; c[idx] = 0.f; }
  if (idx < (size_t)N_NODES * OUTD) theta[idx] = 0.f;
  if (idx < (size_t)N_NODES * FINP) hin[idx] = 0.f;  // zeros pad cols 72..95 forever
}

// pack B (rows x cols f32, row-major) into WMMA fragment lane order
__global__ void pack_b_kernel(const float* __restrict__ W, half_t* __restrict__ frag,
                              int rows, int cols, int ktiles)
{
  int idx = blockIdx.x * blockDim.x + threadIdx.x;
  const int ntiles = cols / 16;
  const int total = ktiles * ntiles * 32 * 16;
  if (idx >= total) return;
  int e    = idx & 15;
  int lane = (idx >> 4) & 31;
  int tile = idx >> 9;  // kt*ntiles + nt
  int kt = tile / ntiles, nt = tile % ntiles;
  int g = (lane >> 4) & 1, n = lane & 15;
  int j = e >> 1, hh = e & 1;
  int kk = kt * 32 + g * 8 + (e & 7) + ((e >> 3) << 4);
  int nn = nt * 16 + n;
  frag[idx] = (kk < rows) ? (half_t)W[(size_t)kk * cols + nn] : (half_t)0.f;
}

__global__ void bsum_kernel(const float* __restrict__ bih, const float* __restrict__ bhh,
                            float* __restrict__ bsum)
{
  int i = threadIdx.x;
  if (i < GATES) bsum[i] = bih[i] + bhh[i];
}

// ---------------- LSTM step (WMMA) -----------------------------------------
__global__ __launch_bounds__(32) void lstm_step_kernel(
    const float* __restrict__ x, const half_t* __restrict__ WihF,
    const half_t* __restrict__ WhhF, const float* __restrict__ bsum,
    float* __restrict__ h, float* __restrict__ c,
    const float* __restrict__ theta, float* __restrict__ hin, int t)
{
  __shared__ float sG[16 * GATES];
  const int lane = threadIdx.x;
  const int m0 = blockIdx.x * 16;
  const int g = (lane >> 4) & 1;
  const int m = lane & 15;
  const int node_a = m0 + m;
  const bool va = node_a < N_NODES;

  const v16h ax  = load_a_rows(x + ((size_t)node_a * T_STEPS + t) * INDIM + g * 8, va);
  const v16h ah0 = load_a_rows(h + (size_t)node_a * FEAT + g * 8, va);
  const v16h ah1 = load_a_rows(h + (size_t)node_a * FEAT + 32 + g * 8, va);
  const int nn = lane & 15;

#pragma unroll 4
  for (int nt = 0; nt < GATES / 16; ++nt) {
    const int n0 = nt * 16;
    __builtin_prefetch(WihF + ((((size_t)(nt + 1) & 15) * 32 + lane) << 4), 0, 1);
    const float bvv = bsum[n0 + nn];
    v8f acc;
#pragma unroll
    for (int r = 0; r < 8; ++r) acc[r] = bvv;
    acc = __builtin_amdgcn_wmma_f32_16x16x32_f16(false, ax, false,
          load_b_frag(WihF, 16, 0, nt, lane), (short)0, acc, false, false);
    acc = __builtin_amdgcn_wmma_f32_16x16x32_f16(false, ah0, false,
          load_b_frag(WhhF, 16, 0, nt, lane), (short)0, acc, false, false);
    acc = __builtin_amdgcn_wmma_f32_16x16x32_f16(false, ah1, false,
          load_b_frag(WhhF, 16, 1, nt, lane), (short)0, acc, false, false);
#pragma unroll
    for (int r = 0; r < 8; ++r) sG[(r + 8 * g) * GATES + n0 + nn] = acc[r];
  }
  __syncthreads();

  // elementwise LSTM cell: gates order i, f, g, o
  for (int idx = lane; idx < 16 * FEAT; idx += 32) {
    int r = idx / FEAT, f = idx % FEAT;
    int node = m0 + r;
    if (node >= N_NODES) continue;
    float gi = sG[r * GATES + f];
    float gf = sG[r * GATES + FEAT + f];
    float gg = sG[r * GATES + 2 * FEAT + f];
    float go = sG[r * GATES + 3 * FEAT + f];
    float cp = c[(size_t)node * FEAT + f];
    float cn = sigmf(gf) * cp + sigmf(gi) * fast_tanh(gg);
    float hn = sigmf(go) * fast_tanh(cn);
    c[(size_t)node * FEAT + f] = cn;
    h[(size_t)node * FEAT + f] = hn;
    hin[(size_t)node * FINP + f] = hn;
  }
  // concat theta into h_in cols 64..71 (cols 72..95 stay zero)
  for (int idx = lane; idx < 16 * OUTD; idx += 32) {
    int r = idx / OUTD, j = idx % OUTD;
    int node = m0 + r;
    if (node >= N_NODES) continue;
    hin[(size_t)node * FINP + FEAT + j] = theta[(size_t)node * OUTD + j];
  }
}

// ---------------- q/k/v/skip projection (WMMA) -----------------------------
__global__ __launch_bounds__(32) void proj_kernel(
    const float* __restrict__ hin,
    const half_t* __restrict__ WqF, const float* __restrict__ bq,
    const half_t* __restrict__ WkF, const float* __restrict__ bk,
    const half_t* __restrict__ WvF, const float* __restrict__ bv,
    const half_t* __restrict__ WsF, const float* __restrict__ bs,
    float* __restrict__ q, float* __restrict__ k, float* __restrict__ v,
    float* __restrict__ skip)
{
  const int lane = threadIdx.x;
  const int m0 = blockIdx.x * 16;
  const int g = (lane >> 4) & 1;
  const int m = lane & 15;
  const int node_a = m0 + m;
  const bool va = node_a < N_NODES;
  const float* rowp = hin + (size_t)node_a * FINP;

  const v16h a0 = load_a_rows(rowp + g * 8, va);
  const v16h a1 = load_a_rows(rowp + 32 + g * 8, va);
  const v16h a2 = load_a_rows(rowp + 64 + g * 8, va);  // cols 72..95 are zero pad
  const int nn = lane & 15;

  const half_t* Wmat[4] = { WqF, WkF, WvF, WsF };
  const float*  Bias[4] = { bq, bk, bv, bs };
  float*        Outp[4] = { q, k, v, skip };
  const int     Cols[4] = { QKV, QKV, QKV, HIDD };

#pragma unroll
  for (int w = 0; w < 4; ++w) {
    const half_t* W = Wmat[w];
    const float* bb = Bias[w];
    float* o = Outp[w];
    const int cols = Cols[w];
    const int ntiles = cols / 16;
#pragma unroll 4
    for (int nt = 0; nt < ntiles; ++nt) {
      const int n0 = nt * 16;
      __builtin_prefetch(W + ((((size_t)2 * ntiles + nt) * 32 + lane) << 4), 0, 1);
      const float bvv = bb[n0 + nn];
      v8f acc;
#pragma unroll
      for (int r = 0; r < 8; ++r) acc[r] = bvv;
      acc = __builtin_amdgcn_wmma_f32_16x16x32_f16(false, a0, false,
            load_b_frag(W, ntiles, 0, nt, lane), (short)0, acc, false, false);
      acc = __builtin_amdgcn_wmma_f32_16x16x32_f16(false, a1, false,
            load_b_frag(W, ntiles, 1, nt, lane), (short)0, acc, false, false);
      acc = __builtin_amdgcn_wmma_f32_16x16x32_f16(false, a2, false,
            load_b_frag(W, ntiles, 2, nt, lane), (short)0, acc, false, false);
#pragma unroll
      for (int r = 0; r < 8; ++r) {
        int node = m0 + r + 8 * g;
        if (node < N_NODES) o[(size_t)node * cols + n0 + nn] = acc[r];
      }
    }
  }
}

// ---------------- edge attention phases ------------------------------------
__global__ void reset_kernel(float* __restrict__ agg, float* __restrict__ mbuf,
                             float* __restrict__ denom)
{
  size_t idx = (size_t)blockIdx.x * blockDim.x + threadIdx.x;
  if (idx < (size_t)N_NODES * QKV) agg[idx] = 0.f;
  if (idx < (size_t)N_NODES * HEADS) { mbuf[idx] = -INFINITY; denom[idx] = 0.f; }
}

__global__ void edge_alpha_kernel(const float* __restrict__ q, const float* __restrict__ k,
                                  const int* __restrict__ src, const int* __restrict__ dst,
                                  float* __restrict__ alpha, float* __restrict__ mbuf)
{
  size_t idx = (size_t)blockIdx.x * blockDim.x + threadIdx.x;
  if (idx >= (size_t)NEDGE * HEADS) return;
  int e = (int)(idx >> 2), hh = (int)(idx & 3);
  int s = src[e], d = dst[e];
  const float* qp = q + (size_t)d * QKV + hh * HIDD;
  const float* kp = k + (size_t)s * QKV + hh * HIDD;
  float acc = 0.f;
#pragma unroll 8
  for (int i = 0; i < HIDD; ++i) acc = fmaf(qp[i], kp[i], acc);
  acc *= 0.125f;  // 1/sqrt(HID)
  alpha[idx] = acc;
  atomicMaxF(mbuf + (size_t)d * HEADS + hh, acc);
}

__global__ void edge_exp_kernel(const int* __restrict__ dst, const float* __restrict__ mbuf,
                                float* __restrict__ alpha, float* __restrict__ denom)
{
  size_t idx = (size_t)blockIdx.x * blockDim.x + threadIdx.x;
  if (idx >= (size_t)NEDGE * HEADS) return;
  int e = (int)(idx >> 2), hh = (int)(idx & 3);
  int d = dst[e];
  float ex = __expf(alpha[idx] - mbuf[(size_t)d * HEADS + hh]);
  alpha[idx] = ex;  // reuse buffer for ex
  atomicAdd(denom + (size_t)d * HEADS + hh, ex);
}

__global__ void edge_agg_kernel(const int* __restrict__ src, const int* __restrict__ dst,
                                const float* __restrict__ alpha, const float* __restrict__ denom,
                                const float* __restrict__ v, float* __restrict__ agg)
{
  size_t idx = (size_t)blockIdx.x * blockDim.x + threadIdx.x;
  if (idx >= (size_t)NEDGE * HEADS) return;
  int e = (int)(idx >> 2), hh = (int)(idx & 3);
  int s = src[e], d = dst[e];
  float w = alpha[idx] / fmaxf(denom[(size_t)d * HEADS + hh], 1e-16f);
  const float* vp = v + (size_t)s * QKV + hh * HIDD;
  float* ap = agg + (size_t)d * QKV + hh * HIDD;
#pragma unroll 4
  for (int i = 0; i < HIDD; ++i) atomicAdd(ap + i, w * vp[i]);
}

// ---------------- per-node update + output transform -----------------------
__global__ void node_update_kernel(const float* __restrict__ agg, const float* __restrict__ skip,
                                   const float* __restrict__ Wmlp, float* __restrict__ theta,
                                   float* __restrict__ out, int t)
{
  int n = blockIdx.x * blockDim.x + threadIdx.x;
  if (n >= N_NODES) return;
  float th[OUTD];
#pragma unroll
  for (int o = 0; o < OUTD; ++o) th[o] = 0.f;
  const float* ag = agg + (size_t)n * QKV;
  const float* sk = skip + (size_t)n * HIDD;
  for (int j = 0; j < HIDD; ++j) {
    float hv = 0.25f * (ag[j] + ag[HIDD + j] + ag[2 * HIDD + j] + ag[3 * HIDD + j]) + sk[j];
    float tv = fast_tanh(hv);
#pragma unroll
    for (int o = 0; o < OUTD; ++o) th[o] += tv * Wmlp[j * OUTD + o];
  }
#pragma unroll
  for (int o = 0; o < OUTD; ++o) theta[(size_t)n * OUTD + o] = th[o];
  float* op = out + ((size_t)n * T_STEPS + t) * OUTD;
#pragma unroll
  for (int o = 0; o < 5; ++o) op[o] = th[o];
  float s5 = sigmf(th[5]), s6 = sigmf(th[6]);
  float a = s5 * s6;
  op[5] = a;
  op[6] = s5 - a;
  float xx = th[7];
  op[7] = fmaxf(xx, 0.f) + log1pf(__expf(-fabsf(xx)));  // stable softplus
}

// ---------------- host-side orchestration ----------------------------------
static inline size_t rup(size_t b) { return (b + 255) & ~(size_t)255; }

extern "C" void kernel_launch(void* const* d_in, const int* in_sizes, int n_in,
                              void* d_out, int out_size, void* d_ws, size_t ws_size,
                              hipStream_t stream) {
  (void)in_sizes; (void)n_in; (void)out_size; (void)ws_size;
  const float* x    = (const float*)d_in[0];
  const int*   ei   = (const int*)d_in[1];
  const float* Wih  = (const float*)d_in[2];
  const float* Whh  = (const float*)d_in[3];
  const float* bih  = (const float*)d_in[4];
  const float* bhh  = (const float*)d_in[5];
  const float* Wq   = (const float*)d_in[6];
  const float* bq   = (const float*)d_in[7];
  const float* Wk   = (const float*)d_in[8];
  const float* bk   = (const float*)d_in[9];
  const float* Wv   = (const float*)d_in[10];
  const float* bv   = (const float*)d_in[11];
  const float* Wsk  = (const float*)d_in[12];
  const float* bsk  = (const float*)d_in[13];
  const float* Wmlp = (const float*)d_in[14];
  const int* src = ei;
  const int* dst = ei + NEDGE;
  float* out = (float*)d_out;

  // workspace bump allocator
  char* ws = (char*)d_ws;
  size_t off = 0;
  auto allocF = [&](size_t n) -> float* { float* p = (float*)(ws + off); off += rup(n * 4); return p; };
  auto allocH = [&](size_t n) -> half_t* { half_t* p = (half_t*)(ws + off); off += rup(n * 2); return p; };

  float* h     = allocF((size_t)N_NODES * FEAT);
  float* c     = allocF((size_t)N_NODES * FEAT);
  float* theta = allocF((size_t)N_NODES * OUTD);
  float* hin   = allocF((size_t)N_NODES * FINP);  // padded stride 96
  float* qb    = allocF((size_t)N_NODES * QKV);
  float* kb    = allocF((size_t)N_NODES * QKV);
  float* vb    = allocF((size_t)N_NODES * QKV);
  float* skip  = allocF((size_t)N_NODES * HIDD);
  float* alpha = allocF((size_t)NEDGE * HEADS);
  float* mbuf  = allocF((size_t)N_NODES * HEADS);
  float* denom = allocF((size_t)N_NODES * HEADS);
  float* agg   = allocF((size_t)N_NODES * QKV);
  float* bsum  = allocF(GATES);
  half_t* WihF = allocH((size_t)1 * 16 * 512);  // ktiles*ntiles*32*16
  half_t* WhhF = allocH((size_t)2 * 16 * 512);
  half_t* WqF  = allocH((size_t)3 * 16 * 512);
  half_t* WkF  = allocH((size_t)3 * 16 * 512);
  half_t* WvF  = allocH((size_t)3 * 16 * 512);
  half_t* WsF  = allocH((size_t)3 * 4 * 512);

  // setup (re-run each call for determinism)
  init_state_kernel<<<(N_NODES * FINP + 255) / 256, 256, 0, stream>>>(h, c, theta, hin);
  pack_b_kernel<<<(1 * 16 * 512 + 255) / 256, 256, 0, stream>>>(Wih, WihF, INDIM, GATES, 1);
  pack_b_kernel<<<(2 * 16 * 512 + 255) / 256, 256, 0, stream>>>(Whh, WhhF, FEAT, GATES, 2);
  pack_b_kernel<<<(3 * 16 * 512 + 255) / 256, 256, 0, stream>>>(Wq, WqF, FIN, QKV, 3);
  pack_b_kernel<<<(3 * 16 * 512 + 255) / 256, 256, 0, stream>>>(Wk, WkF, FIN, QKV, 3);
  pack_b_kernel<<<(3 * 16 * 512 + 255) / 256, 256, 0, stream>>>(Wv, WvF, FIN, QKV, 3);
  pack_b_kernel<<<(3 * 4 * 512 + 255) / 256, 256, 0, stream>>>(Wsk, WsF, FIN, HIDD, 3);
  bsum_kernel<<<1, 256, 0, stream>>>(bih, bhh, bsum);

  const int egrid = (int)(((size_t)NEDGE * HEADS + 255) / 256);
  const int rgrid = (int)(((size_t)N_NODES * QKV + 255) / 256);

  for (int t = 0; t < T_STEPS; ++t) {
    lstm_step_kernel<<<NTILES, 32, 0, stream>>>(x, WihF, WhhF, bsum, h, c, theta, hin, t);
    proj_kernel<<<NTILES, 32, 0, stream>>>(hin, WqF, bq, WkF, bk, WvF, bv, WsF, bsk,
                                           qb, kb, vb, skip);
    reset_kernel<<<rgrid, 256, 0, stream>>>(agg, mbuf, denom);
    edge_alpha_kernel<<<egrid, 256, 0, stream>>>(qb, kb, src, dst, alpha, mbuf);
    edge_exp_kernel<<<egrid, 256, 0, stream>>>(dst, mbuf, alpha, denom);
    edge_agg_kernel<<<egrid, 256, 0, stream>>>(src, dst, alpha, denom, vb, agg);
    node_update_kernel<<<(N_NODES + 255) / 256, 256, 0, stream>>>(agg, skip, Wmlp, theta, out, t);
  }
}
